// Mixer_3504693314043
// MI455X (gfx1250) — compile-verified, matrix operand order checked by reading, not run on hardware
//
#include <hip/hip_runtime.h>

// ---------------- types ----------------
typedef __attribute__((ext_vector_type(16))) __bf16 v16bf;
typedef __attribute__((ext_vector_type(8)))  float  v8f;

__device__ __forceinline__ __bf16 f2bf(float f) {
    unsigned u = __builtin_bit_cast(unsigned, f);
    unsigned r = u + 0x7FFFu + ((u >> 16) & 1u);
    unsigned short h = (unsigned short)(r >> 16);
    return __builtin_bit_cast(__bf16, h);
}
__device__ __forceinline__ float bf2f(__bf16 b) {
    unsigned short s = __builtin_bit_cast(unsigned short, b);
    unsigned u = ((unsigned)s) << 16;
    return __builtin_bit_cast(float, u);
}
__device__ __forceinline__ v8f wmma_bf16(v16bf a, v16bf b, v8f c) {
    return __builtin_amdgcn_wmma_f32_16x16x32_bf16(false, a, false, b, (short)0, c, false, false);
}

// async global->LDS copy of 16 bytes (CDNA5 GLOBAL_LOAD_ASYNC_TO_LDS_B128, ASYNCcnt)
__device__ __forceinline__ void async_b128(unsigned lds_addr, const void* gaddr) {
    asm volatile("global_load_async_to_lds_b128 %0, %1, off"
                 :: "v"(lds_addr), "v"((unsigned long)gaddr) : "memory");
}
__device__ __forceinline__ void async_b128_off16(unsigned lds_addr, const void* gaddr) {
    asm volatile("global_load_async_to_lds_b128 %0, %1, off offset:16"
                 :: "v"(lds_addr), "v"((unsigned long)gaddr) : "memory");
}
__device__ __forceinline__ void wait_async0() {
    asm volatile("s_wait_asynccnt 0x0" ::: "memory");
}
__device__ __forceinline__ unsigned lds_off32(const void* p) {
    return (unsigned)(unsigned long)p;   // generic LDS ptr: low 32 bits are the LDS byte address
}

// ---------------- problem constants ----------------
constexpr int BATCH = 2;
constexpr int L     = 4096;
constexpr int BL    = BATCH * L;        // 8192
constexpr int DM    = 2048;             // d_model
constexpr int DI    = 2048;             // d_inner
constexpr int HN    = 32;               // heads (n_qk == n_v)
constexpr int PDIM  = 64;               // d_inner / n_v
constexpr int NST   = 64;               // d_state
constexpr int CONVD = 6144;             // conv channels
constexpr int IO    = 8224;             // in-proj out dim
constexpr int CH    = 128;              // chunk
constexpr int NC    = L / CH;           // 32 chunks / batch

// ---------------- fp32 -> bf16 convert ----------------
__global__ __launch_bounds__(256) void k_cvt(const float* __restrict__ s,
                                             __bf16* __restrict__ d, long n) {
    long i = (long)blockIdx.x * 256 + threadIdx.x;
    long stride = (long)gridDim.x * 256;
    for (; i < n; i += stride) d[i] = f2bf(s[i]);
}

// ---------------- generic bf16 WMMA GEMM, async double-buffered ----------------
// C[M x N] (f32) = A[M x K] (bf16, row-major) * Bw[N x K]^T (bf16, row-major)
// block = 256 threads (8 waves), tile 128x128, K-step 32
__global__ __launch_bounds__(256) void k_gemm(const __bf16* __restrict__ A,
                                              const __bf16* __restrict__ Bw,
                                              float* __restrict__ C,
                                              int M, int N, int K) {
    __shared__ alignas(32) __bf16 As[2][128][48];
    __shared__ alignas(32) __bf16 Bs[2][128][48];
    const int tid  = threadIdx.x;
    const int lane = tid & 31;
    const int wave = tid >> 5;
    const int wm = wave & 3;      // 4 waves along M -> 32 rows each
    const int wn = wave >> 2;     // 2 waves along N -> 64 cols each
    const int m0 = blockIdx.y * 128;
    const int n0 = blockIdx.x * 128;
    const int r    = lane & 15;
    const int half = lane >> 4;

    v8f acc[2][4];
#pragma unroll
    for (int i = 0; i < 2; ++i)
#pragma unroll
        for (int j = 0; j < 4; ++j) acc[i][j] = (v8f)0.0f;

    const int lrow = tid >> 1;          // 0..127
    const int lcol = (tid & 1) * 16;    // 0 or 16
    const bool bvalid = (n0 + lrow) < N;

    // pre-zero out-of-range B rows (persist across k-steps; asyncs never touch them)
    if (!bvalid) {
#pragma unroll
        for (int i = 0; i < 16; ++i) {
            Bs[0][lrow][lcol + i] = f2bf(0.0f);
            Bs[1][lrow][lcol + i] = f2bf(0.0f);
        }
    }

    const __bf16* aG = A  + (long)(m0 + lrow) * K + lcol;
    const __bf16* bG = Bw + (long)(n0 + lrow) * K + lcol;
    const unsigned laA0 = lds_off32(&As[0][lrow][lcol]);
    const unsigned laA1 = lds_off32(&As[1][lrow][lcol]);
    const unsigned laB0 = lds_off32(&Bs[0][lrow][lcol]);
    const unsigned laB1 = lds_off32(&Bs[1][lrow][lcol]);

    // prologue: stage tile 0 into buffer 0
    async_b128(laA0, aG);
    async_b128_off16(laA0, aG);
    if (bvalid) {
        async_b128(laB0, bG);
        async_b128_off16(laB0, bG);
    }
    wait_async0();
    __syncthreads();

    const int nk = K >> 5;
    for (int ki = 0; ki < nk; ++ki) {
        const int cur = ki & 1;
        // prefetch next tile into the other buffer (just vacated by last iteration)
        if (ki + 1 < nk) {
            const __bf16* aN = aG + (ki + 1) * 32;
            const __bf16* bN = bG + (ki + 1) * 32;
            unsigned la = cur ? laA0 : laA1;
            unsigned lb = cur ? laB0 : laB1;
            async_b128(la, aN);
            async_b128_off16(la, aN);
            if (bvalid) {
                async_b128(lb, bN);
                async_b128_off16(lb, bN);
            }
        }
        // compute on current buffer
        v16bf va[2], vb[4];
#pragma unroll
        for (int i = 0; i < 2; ++i)
            va[i] = *(const v16bf*)&As[cur][wm * 32 + i * 16 + r][half * 16];
#pragma unroll
        for (int j = 0; j < 4; ++j)
            vb[j] = *(const v16bf*)&Bs[cur][wn * 64 + j * 16 + r][half * 16];
#pragma unroll
        for (int i = 0; i < 2; ++i)
#pragma unroll
            for (int j = 0; j < 4; ++j)
                acc[i][j] = wmma_bf16(va[i], vb[j], acc[i][j]);

        wait_async0();
        __syncthreads();
    }

#pragma unroll
    for (int i = 0; i < 2; ++i)
#pragma unroll
        for (int j = 0; j < 4; ++j)
#pragma unroll
            for (int rr = 0; rr < 8; ++rr) {
                int mrow = m0 + wm * 32 + i * 16 + rr + half * 8;
                int ncol = n0 + wn * 64 + j * 16 + (lane & 15);
                if (ncol < N) C[(long)mrow * N + ncol] = acc[i][j][rr];
            }
}

// ---------------- depthwise causal conv (K=4) + split ----------------
__global__ __launch_bounds__(256) void k_conv(const float* __restrict__ xbcza,
                                              const float* __restrict__ cw,
                                              const float* __restrict__ cb,
                                              float* __restrict__ x_f,
                                              __bf16* __restrict__ x_bf,
                                              __bf16* __restrict__ b_bf,
                                              __bf16* __restrict__ c_bf) {
    const int bl = blockIdx.x;
    const int c  = blockIdx.y * 256 + threadIdx.x;   // < 6144
    const int b  = bl / L, l = bl % L;
    float acc = cb[c];
    const float w0 = cw[c * 4 + 0], w1 = cw[c * 4 + 1];
    const float w2 = cw[c * 4 + 2], w3 = cw[c * 4 + 3];
#pragma unroll
    for (int k = 0; k < 4; ++k) {
        int ll = l - 3 + k;
        if (ll >= 0) {
            float wv = (k == 0) ? w0 : (k == 1) ? w1 : (k == 2) ? w2 : w3;
            acc += xbcza[(long)(b * L + ll) * IO + c] * wv;
        }
    }
    if (c < DI) {
        x_f[(long)bl * DI + c]  = acc;
        x_bf[(long)bl * DI + c] = f2bf(acc);
    } else if (c < DI + HN * NST) {
        b_bf[(long)bl * (HN * NST) + (c - DI)] = f2bf(acc);
    } else {
        c_bf[(long)bl * (HN * NST) + (c - DI - HN * NST)] = f2bf(acc);
    }
}

// ---------------- softplus -> dt ----------------
__global__ __launch_bounds__(256) void k_dt(const float* __restrict__ xbcza,
                                            float* __restrict__ dt) {
    long idx = (long)blockIdx.x * 256 + threadIdx.x;   // bl*32 + h
    long bl = idx >> 5;
    int  h  = (int)(idx & 31);
    float a = xbcza[bl * IO + (CONVD + DI) + h];
    dt[idx] = (a > 20.0f) ? a : log1pf(__expf(a));
}

// ---------------- per-chunk cumsum of dt ----------------
__global__ __launch_bounds__(128) void k_cumsum(const float* __restrict__ dt,
                                                float* __restrict__ cumb,
                                                float* __restrict__ cdec) {
    __shared__ float s[CH];
    const int bid = blockIdx.x;                 // (b*NC + c)*HN + h
    const int h   = bid & (HN - 1);
    const int bc  = bid / HN;
    const int b   = bc / NC, c = bc % NC;
    const long base_bl = (long)b * L + (long)c * CH;
    const int q = threadIdx.x;
    s[q] = dt[(base_bl + q) * HN + h];
    __syncthreads();
    for (int off = 1; off < CH; off <<= 1) {
        float add = (q >= off) ? s[q - off] : 0.0f;
        __syncthreads();
        s[q] += add;
        __syncthreads();
    }
    cumb[(long)bid * CH + q] = s[q];
    if (q == CH - 1) cdec[bid] = __expf(s[CH - 1]);
}

// ---------------- chunk states: S[p][n] = sum_k x[k][p] * B[k][n]*exp(cumL-cum[k]) ----------------
__global__ __launch_bounds__(256) void k_states(const __bf16* __restrict__ x_bf,
                                                const __bf16* __restrict__ b_bf,
                                                const float* __restrict__ cumb,
                                                float* __restrict__ states) {
    __shared__ alignas(32) __bf16 sXT[PDIM][144];   // x^T  [p][k]
    __shared__ alignas(32) __bf16 sBT[NST][144];    // (B*sd)^T [n][k]
    __shared__ float sC[CH];
    const int bid = blockIdx.x;
    const int h   = bid & (HN - 1);
    const int bc  = bid / HN;
    const int b   = bc / NC, c = bc % NC;
    const long base_bl = (long)b * L + (long)c * CH;
    const int tid = threadIdx.x;
    if (tid < CH) sC[tid] = cumb[(long)bid * CH + tid];
    __syncthreads();
    const float cl = sC[CH - 1];
    for (int i = tid; i < PDIM * CH; i += 256) {
        int p = i & (PDIM - 1);
        int k = i >> 6;
        sXT[p][k] = x_bf[(base_bl + k) * DI + h * PDIM + p];
        float sd = __expf(cl - sC[k]);
        sBT[p][k] = f2bf(bf2f(b_bf[(base_bl + k) * (HN * NST) + h * NST + p]) * sd);
    }
    __syncthreads();

    const int lane = tid & 31, wave = tid >> 5;
    const int wm = wave & 3;      // p tile (16 rows)
    const int wn = wave >> 2;     // n half (32 cols)
    const int r = lane & 15, half = lane >> 4;
    v8f acc[2];
    acc[0] = (v8f)0.0f; acc[1] = (v8f)0.0f;
#pragma unroll
    for (int ks = 0; ks < 4; ++ks) {
        int k0 = ks * 32;
        v16bf a = *(const v16bf*)&sXT[wm * 16 + r][k0 + half * 16];
#pragma unroll
        for (int jt = 0; jt < 2; ++jt) {
            v16bf bb = *(const v16bf*)&sBT[wn * 32 + jt * 16 + r][k0 + half * 16];
            acc[jt] = wmma_bf16(a, bb, acc[jt]);
        }
    }
#pragma unroll
    for (int jt = 0; jt < 2; ++jt)
#pragma unroll
        for (int rr = 0; rr < 8; ++rr) {
            int p = wm * 16 + rr + half * 8;
            int n = wn * 32 + jt * 16 + (lane & 15);
            states[(long)bid * (PDIM * NST) + p * NST + n] = acc[jt][rr];
        }
}

// ---------------- intra-chunk: CB = C*B^T, decay-mask, y_intra = CBd*x, + D*x ----------------
__global__ __launch_bounds__(256) void k_intra(const __bf16* __restrict__ c_bf,
                                               const __bf16* __restrict__ b_bf,
                                               const __bf16* __restrict__ x_bf,
                                               const float* __restrict__ x_f,
                                               const float* __restrict__ cumb,
                                               const float* __restrict__ Dv,
                                               float* __restrict__ yacc) {
    __shared__ alignas(32) __bf16 sCB[CH][144];     // masked/decayed CB (bf16)
    __shared__ alignas(32) __bf16 sXT[PDIM][144];   // x^T [p][k]
    __shared__ float sC[CH];
    const int bid = blockIdx.x;
    const int h   = bid & (HN - 1);
    const int bc  = bid / HN;
    const int b   = bc / NC, c = bc % NC;
    const long base_bl = (long)b * L + (long)c * CH;
    const int tid = threadIdx.x;
    if (tid < CH) sC[tid] = cumb[(long)bid * CH + tid];
    for (int i = tid; i < PDIM * CH; i += 256) {
        int p = i & (PDIM - 1);
        int k = i >> 6;
        sXT[p][k] = x_bf[(base_bl + k) * DI + h * PDIM + p];
    }
    __syncthreads();

    const int lane = tid & 31, wave = tid >> 5;
    const int wm = wave & 3;     // q: 32 rows per wave
    const int wn = wave >> 2;    // k-col: 64 per wave (for CB) / p: 32 per wave (for y)
    const int r = lane & 15, half = lane >> 4;

    // ---- CB = C (128x64) * B^T (64x128) ----
    v8f acc[2][4];
#pragma unroll
    for (int i = 0; i < 2; ++i)
#pragma unroll
        for (int j = 0; j < 4; ++j) acc[i][j] = (v8f)0.0f;
#pragma unroll
    for (int ks = 0; ks < 2; ++ks) {               // K = 64 (n-state)
        int k0 = ks * 32;
        v16bf va[2], vb[4];
#pragma unroll
        for (int it = 0; it < 2; ++it) {
            int q = wm * 32 + it * 16 + r;
            va[it] = *(const v16bf*)(c_bf + (base_bl + q) * (HN * NST) + h * NST + k0 + half * 16);
        }
#pragma unroll
        for (int jt = 0; jt < 4; ++jt) {
            int kk = wn * 64 + jt * 16 + r;        // B row (l-index within chunk)
            vb[jt] = *(const v16bf*)(b_bf + (base_bl + kk) * (HN * NST) + h * NST + k0 + half * 16);
        }
#pragma unroll
        for (int it = 0; it < 2; ++it)
#pragma unroll
            for (int jt = 0; jt < 4; ++jt)
                acc[it][jt] = wmma_bf16(va[it], vb[jt], acc[it][jt]);
    }
    // decay+mask, write bf16 to LDS
#pragma unroll
    for (int it = 0; it < 2; ++it)
#pragma unroll
        for (int jt = 0; jt < 4; ++jt)
#pragma unroll
            for (int rr = 0; rr < 8; ++rr) {
                int q  = wm * 32 + it * 16 + rr + half * 8;
                int kk = wn * 64 + jt * 16 + (lane & 15);
                float v = (kk <= q) ? acc[it][jt][rr] * __expf(sC[q] - sC[kk]) : 0.0f;
                sCB[q][kk] = f2bf(v);
            }
    __syncthreads();

    // ---- y_intra = CBd (128x128) * x (128x64), + D[h]*x ----
    v8f acc2[2][2];
#pragma unroll
    for (int i = 0; i < 2; ++i)
#pragma unroll
        for (int j = 0; j < 2; ++j) acc2[i][j] = (v8f)0.0f;
#pragma unroll
    for (int ks = 0; ks < 4; ++ks) {               // K = 128
        int k0 = ks * 32;
        v16bf va[2], vb[2];
#pragma unroll
        for (int it = 0; it < 2; ++it)
            va[it] = *(const v16bf*)&sCB[wm * 32 + it * 16 + r][k0 + half * 16];
#pragma unroll
        for (int jt = 0; jt < 2; ++jt)
            vb[jt] = *(const v16bf*)&sXT[wn * 32 + jt * 16 + r][k0 + half * 16];
#pragma unroll
        for (int it = 0; it < 2; ++it)
#pragma unroll
            for (int jt = 0; jt < 2; ++jt)
                acc2[it][jt] = wmma_bf16(va[it], vb[jt], acc2[it][jt]);
    }
    const float dd = Dv[h];
#pragma unroll
    for (int it = 0; it < 2; ++it)
#pragma unroll
        for (int jt = 0; jt < 2; ++jt)
#pragma unroll
            for (int rr = 0; rr < 8; ++rr) {
                int q = wm * 32 + it * 16 + rr + half * 8;
                int p = wn * 32 + jt * 16 + (lane & 15);
                long gi = (base_bl + q) * DI + h * PDIM + p;
                yacc[gi] = acc2[it][jt][rr] + dd * x_f[gi];
            }
}

// ---------------- inter-chunk sequential scan ----------------
__global__ __launch_bounds__(256) void k_scan(const float* __restrict__ states,
                                              const float* __restrict__ cdec,
                                              __bf16* __restrict__ prev_bf) {
    const int bid = blockIdx.x;         // b*HN + h
    const int b = bid >> 5, h = bid & 31;
    const int tid = threadIdx.x;
    float carry[16];
#pragma unroll
    for (int e = 0; e < 16; ++e) carry[e] = 0.0f;
    for (int c = 0; c < NC; ++c) {
        long base = ((long)(b * NC + c) * HN + h) * (PDIM * NST) + tid * 16;
        float cd = cdec[(b * NC + c) * HN + h];
#pragma unroll
        for (int e = 0; e < 16; ++e) prev_bf[base + e] = f2bf(carry[e]);
#pragma unroll
        for (int e = 0; e < 16; ++e) carry[e] = carry[e] * cd + states[base + e];
    }
}

// ---------------- y_inter = exp(cum[q]) * C (128x64) * prev^T (64x64), accumulate ----------------
__global__ __launch_bounds__(256) void k_yinter(const __bf16* __restrict__ c_bf,
                                                const __bf16* __restrict__ prev_bf,
                                                const float* __restrict__ cumb,
                                                float* __restrict__ yacc) {
    const int bid = blockIdx.x;
    const int h   = bid & (HN - 1);
    const int bc  = bid / HN;
    const int b   = bc / NC, c = bc % NC;
    const long base_bl = (long)b * L + (long)c * CH;
    const int tid = threadIdx.x;
    const int lane = tid & 31, wave = tid >> 5;
    const int wm = wave & 3, wn = wave >> 2;
    const int r = lane & 15, half = lane >> 4;
    v8f acc[2][2];
#pragma unroll
    for (int i = 0; i < 2; ++i)
#pragma unroll
        for (int j = 0; j < 2; ++j) acc[i][j] = (v8f)0.0f;
#pragma unroll
    for (int ks = 0; ks < 2; ++ks) {               // K = 64 (n-state)
        int k0 = ks * 32;
        v16bf va[2], vb[2];
#pragma unroll
        for (int it = 0; it < 2; ++it) {
            int q = wm * 32 + it * 16 + r;
            va[it] = *(const v16bf*)(c_bf + (base_bl + q) * (HN * NST) + h * NST + k0 + half * 16);
        }
#pragma unroll
        for (int jt = 0; jt < 2; ++jt) {
            int p = wn * 32 + jt * 16 + r;
            vb[jt] = *(const v16bf*)(prev_bf + (long)bid * (PDIM * NST) + p * NST + k0 + half * 16);
        }
#pragma unroll
        for (int it = 0; it < 2; ++it)
#pragma unroll
            for (int jt = 0; jt < 2; ++jt)
                acc[it][jt] = wmma_bf16(va[it], vb[jt], acc[it][jt]);
    }
#pragma unroll
    for (int it = 0; it < 2; ++it)
#pragma unroll
        for (int jt = 0; jt < 2; ++jt)
#pragma unroll
            for (int rr = 0; rr < 8; ++rr) {
                int q = wm * 32 + it * 16 + rr + half * 8;
                int p = wn * 32 + jt * 16 + (lane & 15);
                float sc = __expf(cumb[(long)bid * CH + q]);
                long gi = (base_bl + q) * DI + h * PDIM + p;
                yacc[gi] += acc[it][jt][rr] * sc;
            }
}

// ---------------- gating: yg = bf16( y * silu(z + z_bias) ) ----------------
__global__ __launch_bounds__(256) void k_gate(const float* __restrict__ yacc,
                                              const float* __restrict__ xbcza,
                                              const float* __restrict__ zb,
                                              __bf16* __restrict__ yg) {
    long idx = (long)blockIdx.x * 256 + threadIdx.x;   // bl*DI + i
    long bl = idx / DI;
    int  i  = (int)(idx % DI);
    float z = xbcza[bl * IO + CONVD + i] + zb[i];
    float s = z / (1.0f + __expf(-z));
    yg[idx] = f2bf(yacc[idx] * s);
}

// ---------------- host launch ----------------
extern "C" void kernel_launch(void* const* d_in, const int* in_sizes, int n_in,
                              void* d_out, int out_size, void* d_ws, size_t ws_size,
                              hipStream_t stream) {
    (void)in_sizes; (void)n_in; (void)out_size; (void)ws_size;
    const float* u      = (const float*)d_in[0];
    const float* W_in   = (const float*)d_in[1];
    const float* conv_w = (const float*)d_in[2];
    const float* conv_b = (const float*)d_in[3];
    const float* Dvec   = (const float*)d_in[4];
    const float* z_bias = (const float*)d_in[5];
    const float* W_out  = (const float*)d_in[6];
    float* out = (float*)d_out;

    char* w = (char*)d_ws;
    size_t off = 0;
    auto alloc = [&](size_t bytes) -> void* {
        void* p = w + off;
        off = (off + bytes + 255) & ~(size_t)255;
        return p;
    };
    __bf16* u_bf    = (__bf16*)alloc((size_t)BL * DM * 2);
    __bf16* win_bf  = (__bf16*)alloc((size_t)IO * DM * 2);
    __bf16* wout_bf = (__bf16*)alloc((size_t)DM * DI * 2);
    float*  xbcza   = (float*) alloc((size_t)BL * IO * 4);
    float*  x_f     = (float*) alloc((size_t)BL * DI * 4);
    __bf16* x_bf    = (__bf16*)alloc((size_t)BL * DI * 2);
    __bf16* b_bf    = (__bf16*)alloc((size_t)BL * HN * NST * 2);
    __bf16* c_bf    = (__bf16*)alloc((size_t)BL * HN * NST * 2);
    float*  dtb     = (float*) alloc((size_t)BL * HN * 4);
    float*  cumb    = (float*) alloc((size_t)BATCH * NC * HN * CH * 4);
    float*  cdec    = (float*) alloc((size_t)BATCH * NC * HN * 4);
    float*  states  = (float*) alloc((size_t)BATCH * NC * HN * PDIM * NST * 4);
    __bf16* prev_bf = (__bf16*)alloc((size_t)BATCH * NC * HN * PDIM * NST * 2);
    float*  yacc    = (float*) alloc((size_t)BL * DI * 4);
    __bf16* yg      = (__bf16*)alloc((size_t)BL * DI * 2);

    // 1) convert to bf16
    k_cvt<<<2048, 256, 0, stream>>>(u, u_bf, (long)BL * DM);
    k_cvt<<<2048, 256, 0, stream>>>(W_in, win_bf, (long)IO * DM);
    k_cvt<<<1024, 256, 0, stream>>>(W_out, wout_bf, (long)DM * DI);

    // 2) input projection GEMM: xBCzA = u * W_in^T  (8192 x 8224, K=2048)
    {
        dim3 g((IO + 127) / 128, BL / 128);
        k_gemm<<<g, 256, 0, stream>>>(u_bf, win_bf, xbcza, BL, IO, DM);
    }

    // 3) depthwise conv + split; softplus
    k_conv<<<dim3(BL, CONVD / 256), 256, 0, stream>>>(xbcza, conv_w, conv_b,
                                                      x_f, x_bf, b_bf, c_bf);
    k_dt<<<(BL * HN) / 256, 256, 0, stream>>>(xbcza, dtb);

    // 4) per-chunk cumsum
    k_cumsum<<<BATCH * NC * HN, CH, 0, stream>>>(dtb, cumb, cdec);

    // 5) chunk states (WMMA 64x64x128)
    k_states<<<BATCH * NC * HN, 256, 0, stream>>>(x_bf, b_bf, cumb, states);

    // 6) intra-chunk (WMMA 128x128x64 -> 128x64x128)
    k_intra<<<BATCH * NC * HN, 256, 0, stream>>>(c_bf, b_bf, x_bf, x_f, cumb,
                                                 Dvec, yacc);

    // 7) inter-chunk sequential scan
    k_scan<<<BATCH * HN, 256, 0, stream>>>(states, cdec, prev_bf);

    // 8) y_inter (WMMA 128x64x64) accumulated into yacc
    k_yinter<<<BATCH * NC * HN, 256, 0, stream>>>(c_bf, prev_bf, cumb, yacc);

    // 9) gate and output GEMM: out = (y*silu(z)) * W_out^T  (8192 x 2048, K=2048)
    k_gate<<<(long)BL * DI / 256, 256, 0, stream>>>(yacc, xbcza, z_bias, yg);
    {
        dim3 g(DM / 128, BL / 128);
        k_gemm<<<g, 256, 0, stream>>>(yg, wout_bf, out, BL, DM, DI);
    }
}
